// MoEBlock_90503550861377
// MI455X (gfx1250) — matrix operand order repaired and checked by reading, not compiled, not run on hardware
//
#include <hip/hip_runtime.h>

// ---------------------------------------------------------------------------
// MI455X (gfx1250) MoE transformer block, bf16 WMMA + async global->LDS.
// Pipeline: residmix+rmsnorm -> QKV GEMMs -> qknorm+rope -> flash attention
//           -> O GEMM -> resid+rmsnorm -> router/top2 -> gather -> expert
//           GEMM1(silu) -> expert GEMM2 -> weighted scatter + residual.
// ---------------------------------------------------------------------------

typedef __bf16 bf16_t;
typedef __attribute__((ext_vector_type(16))) __bf16 v16bf;
typedef __attribute__((ext_vector_type(8)))  __bf16 v8bf;
typedef __attribute__((ext_vector_type(8)))  float  v8f;
typedef __attribute__((ext_vector_type(4)))  float  v4f;
typedef __attribute__((ext_vector_type(4)))  int    v4i;

union BF16x16 { v16bf v; v8bf h[2]; };

// gfx1250 async global->LDS copy path (ASYNCcnt), with safe fallback.
#if defined(__gfx1250__) && __has_builtin(__builtin_amdgcn_global_load_async_to_lds_b128)
#define HAS_ASYNC_LDS 1
#else
#define HAS_ASYNC_LDS 0
#endif

#if HAS_ASYNC_LDS
// Builtin signature: (global v4i* src, lds v4i* dst, imm offset, imm cpol)
typedef __attribute__((address_space(1))) v4i* gv4i_p;
typedef __attribute__((address_space(3))) v4i* lv4i_p;
__device__ __forceinline__ void async_copy16(const void* g, void* l) {
  __builtin_amdgcn_global_load_async_to_lds_b128((gv4i_p)g, (lv4i_p)l, 0, 0);
}
__device__ __forceinline__ void wait_async() {
#if __has_builtin(__builtin_amdgcn_s_wait_asynccnt)
  __builtin_amdgcn_s_wait_asynccnt(0);
#else
  asm volatile("s_wait_asynccnt 0x0" ::: "memory");
#endif
}
#else
__device__ __forceinline__ void wait_async() {}
#endif

__device__ __forceinline__ bf16_t f2bf(float f) { return (bf16_t)f; }

__device__ __forceinline__ v8f vzero() {
  v8f v = {0.f, 0.f, 0.f, 0.f, 0.f, 0.f, 0.f, 0.f};
  return v;
}

__device__ __forceinline__ v8f wmma_bf16(v16bf a, v16bf b, v8f c) {
  // D = A(16x32 bf16) * B(32x16 bf16) + C(16x16 f32)
  return __builtin_amdgcn_wmma_f32_16x16x32_bf16(
      false, a, false, b, (short)0, c, false, false);
}

// A-frag (16x32 bf16) from LDS tile, row-major with stride (bf16 units).
// ISA layout: lane m (&15) holds row m; half hh: elems 0..7 = K[hh*8..],
// elems 8..15 = K[16+hh*8..].
__device__ __forceinline__ v16bf load_afrag(const bf16_t* tile, int mbase,
                                            int stride, int lane) {
  int m = lane & 15, hh = lane >> 4;
  const bf16_t* r = tile + (mbase + m) * stride;
  BF16x16 f;
  f.h[0] = *(const v8bf*)(r + hh * 8);
  f.h[1] = *(const v8bf*)(r + 16 + hh * 8);
  return f.v;
}

// B-frag (32x16 bf16) from an N-major [n][k] LDS tile with row stride.
// ISA layout: lane n (&15) = column n; half hh: elems 0..7 = K[hh*8..],
// elems 8..15 = K[16+hh*8..].
__device__ __forceinline__ v16bf load_bfrag(const bf16_t* tileT, int nbase,
                                            int kbase, int stride, int lane) {
  int n = lane & 15, hh = lane >> 4;
  const bf16_t* r = tileT + (nbase + n) * stride + kbase + hh * 8;
  BF16x16 f;
  f.h[0] = *(const v8bf*)(r);
  f.h[1] = *(const v8bf*)(r + 16);
  return f.v;
}

// ---------------------------------------------------------------------------
// Generic / expert-segmented bf16 WMMA GEMM, double-buffered LDS, async A.
// C[64x128 block] = A(bf16, lda=K) * B(fp32 weights, ldb=N), K%32==0.
// 256 threads = 8 waves; wave computes 16(M)x64(N) via 4 accumulators.
// MODE 0: fp32 store.  MODE 1: silu -> bf16 store.
// pad!=nullptr => expert mode: blockIdx.z = expert, early-exit past pad[e].
// ---------------------------------------------------------------------------
template <int MODE>
__global__ __launch_bounds__(256)
void gemm_bf16_kernel(const bf16_t* __restrict__ A, int lda,
                      const float* __restrict__ B, int ldb,
                      void* __restrict__ Cout, int ldc, int K,
                      const int* __restrict__ pad,
                      long sA, long sB, long sC) {
  int e = blockIdx.z;
  if (pad && (int)(blockIdx.y * 64) >= pad[e]) return;
  A += (long)e * sA;
  B += (long)e * sB;

  __shared__ bf16_t Asub[2][64 * 32];  // [m][k]
  __shared__ bf16_t Bt[2][128 * 32];   // [n][k] transposed, bf16-converted

  int tid = threadIdx.x;
  int lane = tid & 31;
  int wave = tid >> 5;
  int wm = (wave & 3) * 16;   // M sub-tile
  int wn = (wave >> 2) * 64;  // N sub-tile

  long mBlk = (long)blockIdx.y * 64;
  long nBlk = (long)blockIdx.x * 128;

  // per-thread A copy slot: 16B (row ar, col ac)
  int ar = tid >> 2, ac = (tid & 3) * 8;
  const bf16_t* Asrc0 = A + (mBlk + ar) * (long)lda + ac;
  // per-thread B slot: row br (0..31), 16 consecutive n at cb
  int br = tid >> 3, cb = (tid & 7) * 16;
  const float* Bsrc0 = B + (long)br * ldb + nBlk + cb;

  auto issueA = [&](int k0, int buf) {
    const void* src = Asrc0 + k0;
    void* dst = &Asub[buf][ar * 32 + ac];
#if HAS_ASYNC_LDS
    async_copy16(src, dst);
#else
    *(v8bf*)dst = *(const v8bf*)src;
#endif
  };
  auto loadB = [&](int k0, v4f* breg) {
    const float* src = Bsrc0 + (long)k0 * ldb;
#pragma unroll
    for (int i = 0; i < 4; ++i) breg[i] = *(const v4f*)(src + 4 * i);
    __builtin_prefetch(src + 64L * ldb, 0, 0);  // 2 K-tiles ahead (weights)
  };
  auto storeB = [&](int buf, const v4f* breg) {
    bf16_t* bt = Bt[buf];
#pragma unroll
    for (int i = 0; i < 4; ++i) {
      v4f x = breg[i];
      bt[(cb + 4 * i + 0) * 32 + br] = f2bf(x.x);
      bt[(cb + 4 * i + 1) * 32 + br] = f2bf(x.y);
      bt[(cb + 4 * i + 2) * 32 + br] = f2bf(x.z);
      bt[(cb + 4 * i + 3) * 32 + br] = f2bf(x.w);
    }
  };

  v8f acc[4] = {vzero(), vzero(), vzero(), vzero()};
  v4f breg[4];

  // prologue: stage tile 0 into buffer 0
  issueA(0, 0);
  loadB(0, breg);
  storeB(0, breg);

  for (int k0 = 0; k0 < K; k0 += 32) {
    int cur = (k0 >> 5) & 1;
    wait_async();
    __syncthreads();  // buf[cur] ready; everyone done reading buf[1-cur]
    bool more = (k0 + 32) < K;
    if (more) {
      issueA(k0 + 32, 1 - cur);  // overlaps with WMMA below
      loadB(k0 + 32, breg);
    }
    v16bf a = load_afrag(Asub[cur], wm, 32, lane);
#pragma unroll
    for (int j = 0; j < 4; ++j) {
      v16bf b = load_bfrag(Bt[cur], wn + j * 16, 0, 32, lane);
      acc[j] = wmma_bf16(a, b, acc[j]);
    }
    if (more) storeB(1 - cur, breg);  // convert after compute issued
  }

  int n = lane & 15, hh = lane >> 4;
  long mrow = mBlk + wm + hh * 8;
  long ncol = nBlk + wn + n;
  if (MODE == 0) {
    float* C = (float*)Cout + (long)e * sC;
#pragma unroll
    for (int j = 0; j < 4; ++j)
#pragma unroll
      for (int r = 0; r < 8; ++r)
        C[(mrow + r) * (long)ldc + ncol + j * 16] = acc[j][r];
  } else {
    bf16_t* C = (bf16_t*)Cout + (long)e * sC;
#pragma unroll
    for (int j = 0; j < 4; ++j)
#pragma unroll
      for (int r = 0; r < 8; ++r) {
        float x = acc[j][r];
        float s = x / (1.f + __expf(-x));  // silu
        C[(mrow + r) * (long)ldc + ncol + j * 16] = f2bf(s);
      }
  }
}

// ---------------------------------------------------------------------------
// Residual-mix / residual-add + RMSNorm. One block per token.
// mode 0: x' = g0[c]*X + g1[c]*Add   (resid_mix rows)
// mode 1: x' = X + g0[c]*Add         (attn_scale residual)
// Writes x' (fp32) and rmsnorm(x') (bf16).
// ---------------------------------------------------------------------------
__global__ __launch_bounds__(256)
void residnorm_kernel(const float* __restrict__ X, const float* __restrict__ Add,
                      const float* __restrict__ g0, const float* __restrict__ g1,
                      float* __restrict__ Xout, bf16_t* __restrict__ Nout,
                      int mode) {
  long t = blockIdx.x;
  int tid = threadIdx.x;
  float v[4];
  float ss = 0.f;
#pragma unroll
  for (int i = 0; i < 4; ++i) {
    int c = tid + i * 256;
    long idx = t * 1024 + c;
    float xv = (mode == 0) ? (g0[c] * X[idx] + g1[c] * Add[idx])
                           : (X[idx] + g0[c] * Add[idx]);
    v[i] = xv;
    ss += xv * xv;
  }
#pragma unroll
  for (int b = 1; b < 32; b <<= 1) ss += __shfl_xor(ss, b, 32);
  __shared__ float red[8];
  if ((tid & 31) == 0) red[tid >> 5] = ss;
  __syncthreads();
  float tot = 0.f;
#pragma unroll
  for (int i = 0; i < 8; ++i) tot += red[i];
  float inv = rsqrtf(tot * (1.0f / 1024.0f) + 1e-6f);
#pragma unroll
  for (int i = 0; i < 4; ++i) {
    int c = tid + i * 256;
    long idx = t * 1024 + c;
    Xout[idx] = v[i];
    Nout[idx] = f2bf(v[i] * inv);
  }
}

// ---------------------------------------------------------------------------
// Per-head QK RMSNorm * gain + rotate-half RoPE; fp32 -> bf16.
// One block per token; wave w handles q head w; waves 0-1: k heads, 2-3: v.
// ---------------------------------------------------------------------------
__device__ __forceinline__ void qk_head(const float* __restrict__ src,
                                        const float* __restrict__ gain,
                                        bf16_t* __restrict__ dst,
                                        float pos, int lane) {
  float x[4];
  float ss = 0.f;
#pragma unroll
  for (int i = 0; i < 4; ++i) {
    x[i] = src[lane + 32 * i];
    ss += x[i] * x[i];
  }
#pragma unroll
  for (int b = 1; b < 32; b <<= 1) ss += __shfl_xor(ss, b, 32);
  float inv = rsqrtf(ss * (1.0f / 128.0f) + 1e-6f);
#pragma unroll
  for (int i = 0; i < 4; ++i) x[i] = x[i] * inv * gain[lane + 32 * i];
  // rotate-half pairs: (i, i+64) for i in 0..63 -> (x[p], x[p+2])
#pragma unroll
  for (int pr = 0; pr < 2; ++pr) {
    int idx = lane + 32 * pr;  // 0..63
    // inv_freq = 10000^(-idx/64) = exp(-idx * ln(10000)/64)
    float ang = pos * __expf((float)idx * -0.14391156f);
    float sn, cs;
    __sincosf(ang, &sn, &cs);
    float a = x[pr], b = x[pr + 2];
    x[pr] = a * cs - b * sn;
    x[pr + 2] = a * sn + b * cs;
  }
#pragma unroll
  for (int i = 0; i < 4; ++i) dst[lane + 32 * i] = f2bf(x[i]);
}

__global__ __launch_bounds__(256)
void qkrope_kernel(const float* __restrict__ Qf, const float* __restrict__ Kf,
                   const float* __restrict__ Vf, const float* __restrict__ gain,
                   bf16_t* __restrict__ Qb, bf16_t* __restrict__ Kb,
                   bf16_t* __restrict__ Vb) {
  long t = blockIdx.x;
  int lane = threadIdx.x & 31;
  int wave = threadIdx.x >> 5;
  float pos = (float)t;
  qk_head(Qf + t * 1024 + wave * 128, gain, Qb + t * 1024 + wave * 128, pos, lane);
  if (wave < 2)
    qk_head(Kf + t * 256 + wave * 128, gain, Kb + t * 256 + wave * 128, pos, lane);
  if (wave >= 2 && wave < 4) {
    int h = wave - 2;
#pragma unroll
    for (int i = 0; i < 4; ++i) {
      int c = h * 128 + lane + 32 * i;
      Vb[t * 256 + c] = f2bf(Vf[t * 256 + c]);
    }
  }
}

// ---------------------------------------------------------------------------
// Flash-style causal attention, bf16 WMMA. grid=(16 qblocks, 8 heads),
// 128 threads (4 waves x 16 query rows). Key blocks of 32, online softmax.
// GQA: head h uses kv head h>>2. K tile staged via async global->LDS.
// ---------------------------------------------------------------------------
__global__ __launch_bounds__(128)
void attn_kernel(const bf16_t* __restrict__ Qb, const bf16_t* __restrict__ Kb,
                 const bf16_t* __restrict__ Vb, bf16_t* __restrict__ Ob) {
  const int head = blockIdx.y;
  const int qblk = blockIdx.x;
  const int kvh = head >> 2;
  int tid = threadIdx.x, lane = tid & 31, wave = tid >> 5;
  int qbase = qblk * 64 + wave * 16;
  int m = lane & 15, hh = lane >> 4;

  __shared__ bf16_t Kt[32 * 128];    // [key][dh]  (natural: B-frag rows)
  __shared__ bf16_t Vt[128 * 32];    // [dh][key]  (transposed for P@V)
  __shared__ bf16_t Pt[4][16 * 32];  // per-wave P relayout scratch

  // Q fragments (4 dh-chunks of 32), straight from global
  v16bf qf[4];
  {
    const bf16_t* qrow = Qb + (long)(qbase + m) * 1024 + head * 128;
#pragma unroll
    for (int c = 0; c < 4; ++c) {
      BF16x16 f;
      f.h[0] = *(const v8bf*)(qrow + c * 32 + hh * 8);
      f.h[1] = *(const v8bf*)(qrow + c * 32 + 16 + hh * 8);
      qf[c] = f.v;
    }
  }

  v8f o[8];
#pragma unroll
  for (int d = 0; d < 8; ++d) o[d] = vzero();
  float rm[8], rl[8];
#pragma unroll
  for (int r = 0; r < 8; ++r) { rm[r] = -3.0e38f; rl[r] = 0.f; }

  const float sc = 0.08838834764831845f;  // 1/sqrt(128)
  int kend = qblk * 64 + 64;              // causal: keys < end of q block

  for (int kb = 0; kb < kend; kb += 32) {
    __syncthreads();
    {  // K block 32x128 -> LDS (async, natural layout)
      int r = tid >> 2, cb = (tid & 3) * 32;
      const bf16_t* src = Kb + (long)(kb + r) * 256 + kvh * 128 + cb;
      bf16_t* dst = Kt + r * 128 + cb;
#pragma unroll
      for (int i = 0; i < 32; i += 8) {
#if HAS_ASYNC_LDS
        async_copy16(src + i, dst + i);
#else
        *(v8bf*)(dst + i) = *(const v8bf*)(src + i);
#endif
      }
    }
    {  // V block 32x128 -> LDS transposed [dh][key]
      int r = tid >> 2, cb = (tid & 3) * 32;
      const bf16_t* src = Vb + (long)(kb + r) * 256 + kvh * 128 + cb;
#pragma unroll
      for (int i = 0; i < 32; ++i) Vt[(cb + i) * 32 + r] = src[i];
    }
    wait_async();
    __syncthreads();

    // scores: two 16x16 frags (keys kb..+15, kb+16..+31)
    v8f s[2];
    s[0] = vzero();
    s[1] = vzero();
#pragma unroll
    for (int j = 0; j < 2; ++j)
#pragma unroll
      for (int c = 0; c < 4; ++c)
        s[j] = wmma_bf16(qf[c], load_bfrag(Kt, j * 16, c * 32, 128, lane), s[j]);

#pragma unroll
    for (int r = 0; r < 8; ++r) {
      int qrow = qbase + hh * 8 + r;
#pragma unroll
      for (int j = 0; j < 2; ++j) {
        int key = kb + j * 16 + (lane & 15);
        float v = s[j][r] * sc;
        s[j][r] = (key <= qrow) ? v : -3.0e38f;
      }
      float mx = fmaxf(s[0][r], s[1][r]);
#pragma unroll
      for (int b = 1; b < 16; b <<= 1) mx = fmaxf(mx, __shfl_xor(mx, b, 32));
      float mnew = fmaxf(rm[r], mx);
      float p0 = __expf(s[0][r] - mnew);
      float p1 = __expf(s[1][r] - mnew);
      s[0][r] = p0;
      s[1][r] = p1;
      float rs = p0 + p1;
#pragma unroll
      for (int b = 1; b < 16; b <<= 1) rs += __shfl_xor(rs, b, 32);
      float scale = __expf(rm[r] - mnew);
      rl[r] = rl[r] * scale + rs;
      rm[r] = mnew;
#pragma unroll
      for (int d = 0; d < 8; ++d) o[d][r] = o[d][r] * scale;
    }

    // P (C layout) -> A-frag layout via per-wave LDS scratch
    bf16_t* pw = Pt[wave];
#pragma unroll
    for (int r = 0; r < 8; ++r) {
      pw[(hh * 8 + r) * 32 + (lane & 15)] = f2bf(s[0][r]);
      pw[(hh * 8 + r) * 32 + 16 + (lane & 15)] = f2bf(s[1][r]);
    }
    v16bf pa = load_afrag(pw, 0, 32, lane);
#pragma unroll
    for (int d = 0; d < 8; ++d)
      o[d] = wmma_bf16(pa, load_bfrag(Vt, d * 16, 0, 32, lane), o[d]);
  }

  // epilogue: normalize, write bf16
#pragma unroll
  for (int r = 0; r < 8; ++r) {
    float inv = 1.f / rl[r];
    long row = qbase + hh * 8 + r;
    bf16_t* dst = Ob + row * 1024 + head * 128 + (lane & 15);
#pragma unroll
    for (int d = 0; d < 8; ++d) dst[d * 16] = f2bf(o[d][r] * inv);
  }
}

// ---------------------------------------------------------------------------
// Router: logits, softmax, top-2, normalized weights, bucket append.
// ---------------------------------------------------------------------------
__global__ void zero_cnt_kernel(int* cnt) {
  if (threadIdx.x < 8) cnt[threadIdx.x] = 0;
}

__global__ __launch_bounds__(256)
void router_kernel(const bf16_t* __restrict__ N2, const float* __restrict__ RW,
                   int* __restrict__ cnt, int* __restrict__ lists,
                   int* __restrict__ slotmap, float* __restrict__ wts) {
  int t = blockIdx.x * 256 + threadIdx.x;
  float acc[8] = {0.f, 0.f, 0.f, 0.f, 0.f, 0.f, 0.f, 0.f};
  const bf16_t* row = N2 + (long)t * 1024;
  for (int d = 0; d < 1024; ++d) {
    float x = (float)row[d];
#pragma unroll
    for (int e = 0; e < 8; ++e) acc[e] += x * RW[d * 8 + e];
  }
  float mx = acc[0];
#pragma unroll
  for (int e = 1; e < 8; ++e) mx = fmaxf(mx, acc[e]);
#pragma unroll
  for (int e = 0; e < 8; ++e) acc[e] = __expf(acc[e] - mx);
  int i1 = 0;
#pragma unroll
  for (int e = 1; e < 8; ++e)
    if (acc[e] > acc[i1]) i1 = e;
  int i2 = (i1 == 0) ? 1 : 0;
#pragma unroll
  for (int e = 0; e < 8; ++e)
    if (e != i1 && acc[e] > acc[i2]) i2 = e;
  float wn = 1.f / (acc[i1] + acc[i2]);  // softmax denom cancels in ratio
  int s1 = atomicAdd(&cnt[i1], 1);
  int s2 = atomicAdd(&cnt[i2], 1);
  lists[i1 * 1024 + s1] = t;
  lists[i2 * 1024 + s2] = t;
  slotmap[t * 2 + 0] = i1 * 1024 + s1;
  slotmap[t * 2 + 1] = i2 * 1024 + s2;
  wts[t * 2 + 0] = acc[i1] * wn;
  wts[t * 2 + 1] = acc[i2] * wn;
}

__global__ void pad_kernel(const int* __restrict__ cnt, int* __restrict__ pad) {
  if (threadIdx.x < 8) pad[threadIdx.x] = (cnt[threadIdx.x] + 63) & ~63;
}

// Gather tokens into per-expert segments (zero-fill padding rows).
__global__ __launch_bounds__(256)
void gather_kernel(const bf16_t* __restrict__ N2, const int* __restrict__ lists,
                   const int* __restrict__ cnt, const int* __restrict__ pad,
                   bf16_t* __restrict__ Ag) {
  int e = blockIdx.y;
  int base = blockIdx.x * 64;
  if (base >= pad[e]) return;
  int c = cnt[e];
  int tid = threadIdx.x;
  for (int r = 0; r < 64; ++r) {
    int slot = base + r;
    unsigned long long* dst =
        (unsigned long long*)(Ag + ((long)e * 1024 + slot) * 1024);
    if (slot < c) {
      int tok = lists[e * 1024 + slot];
      const unsigned long long* src =
          (const unsigned long long*)(N2 + (long)tok * 1024);
      dst[tid] = src[tid];  // 256 threads x 8B = 2KB row
    } else {
      dst[tid] = 0ULL;
    }
  }
}

// Final: out = x + mlp_scale * (w0*Oe[slot0] + w1*Oe[slot1])
__global__ __launch_bounds__(256)
void final_kernel(const float* __restrict__ Xb, const float* __restrict__ Oe,
                  const int* __restrict__ slotmap, const float* __restrict__ wts,
                  const float* __restrict__ mlp_scale, float* __restrict__ Out) {
  long t = blockIdx.x;
  long s0 = slotmap[t * 2 + 0];
  long s1 = slotmap[t * 2 + 1];
  float w0 = wts[t * 2 + 0];
  float w1 = wts[t * 2 + 1];
#pragma unroll
  for (int i = 0; i < 4; ++i) {
    int c = threadIdx.x + i * 256;
    float h = w0 * Oe[s0 * 1024 + c] + w1 * Oe[s1 * 1024 + c];
    Out[t * 1024 + c] = Xb[t * 1024 + c] + mlp_scale[c] * h;
  }
}

// ---------------------------------------------------------------------------
extern "C" void kernel_launch(void* const* d_in, const int* in_sizes, int n_in,
                              void* d_out, int out_size, void* d_ws,
                              size_t ws_size, hipStream_t stream) {
  (void)in_sizes; (void)n_in; (void)out_size; (void)ws_size;
  const float* x        = (const float*)d_in[0];
  const float* v1       = (const float*)d_in[1];
  const float* wq       = (const float*)d_in[2];
  const float* wk       = (const float*)d_in[3];
  const float* wv       = (const float*)d_in[4];
  const float* wo       = (const float*)d_in[5];
  const float* qk_gain  = (const float*)d_in[6];
  const float* router_w = (const float*)d_in[7];
  const float* w1       = (const float*)d_in[8];
  const float* w2       = (const float*)d_in[9];
  const float* attn_s   = (const float*)d_in[10];
  const float* mlp_s    = (const float*)d_in[11];
  const float* rmix     = (const float*)d_in[12];

  char* base = (char*)d_ws;
  size_t off = 0;
  auto alloc = [&](size_t bytes) -> void* {
    void* r = base + off;
    off += (bytes + 255) & ~(size_t)255;
    return r;
  };

  const long S = 1024, D = 1024;
  float*  xbuf  = (float*)alloc(S * D * 4);
  bf16_t* nbuf  = (bf16_t*)alloc(S * D * 2);
  float*  qf    = (float*)alloc(S * 1024 * 4);
  float*  kf    = (float*)alloc(S * 256 * 4);
  float*  vf    = (float*)alloc(S * 256 * 4);
  bf16_t* qb    = (bf16_t*)alloc(S * 1024 * 2);
  bf16_t* kb    = (bf16_t*)alloc(S * 256 * 2);
  bf16_t* vb    = (bf16_t*)alloc(S * 256 * 2);
  bf16_t* ob    = (bf16_t*)alloc(S * 1024 * 2);
  float*  aproj = (float*)alloc(S * D * 4);
  bf16_t* n2    = (bf16_t*)alloc(S * D * 2);
  int*    cnt   = (int*)alloc(8 * 4);
  int*    padc  = (int*)alloc(8 * 4);
  int*    lists = (int*)alloc(8 * 1024 * 4);
  int*    smap  = (int*)alloc(S * 2 * 4);
  float*  swts  = (float*)alloc(S * 2 * 4);
  bf16_t* Ag    = (bf16_t*)alloc(8L * 1024 * 1024 * 2);   // 16 MB
  bf16_t* Hb    = (bf16_t*)alloc(8L * 1024 * 4096 * 2);   // 64 MB
  float*  Oe    = (float*)alloc(8L * 1024 * 1024 * 4);    // 32 MB

  // 1) resid mix + rmsnorm
  residnorm_kernel<<<1024, 256, 0, stream>>>(x, v1, rmix, rmix + 1024,
                                             xbuf, nbuf, 0);
  // 2) QKV projections (bf16 WMMA)
  gemm_bf16_kernel<0><<<dim3(8, 16, 1), 256, 0, stream>>>(
      nbuf, 1024, wq, 1024, qf, 1024, 1024, nullptr, 0, 0, 0);
  gemm_bf16_kernel<0><<<dim3(2, 16, 1), 256, 0, stream>>>(
      nbuf, 1024, wk, 256, kf, 256, 1024, nullptr, 0, 0, 0);
  gemm_bf16_kernel<0><<<dim3(2, 16, 1), 256, 0, stream>>>(
      nbuf, 1024, wv, 256, vf, 256, 1024, nullptr, 0, 0, 0);
  // 3) QK-norm + RoPE -> bf16
  qkrope_kernel<<<1024, 256, 0, stream>>>(qf, kf, vf, qk_gain, qb, kb, vb);
  // 4) flash causal attention
  attn_kernel<<<dim3(16, 8), 128, 0, stream>>>(qb, kb, vb, ob);
  // 5) output projection
  gemm_bf16_kernel<0><<<dim3(8, 16, 1), 256, 0, stream>>>(
      ob, 1024, wo, 1024, aproj, 1024, 1024, nullptr, 0, 0, 0);
  // 6) residual + rmsnorm for MoE
  residnorm_kernel<<<1024, 256, 0, stream>>>(xbuf, aproj, attn_s, nullptr,
                                             xbuf, n2, 1);
  // 7) routing
  zero_cnt_kernel<<<1, 32, 0, stream>>>(cnt);
  router_kernel<<<4, 256, 0, stream>>>(n2, router_w, cnt, lists, smap, swts);
  pad_kernel<<<1, 32, 0, stream>>>(cnt, padc);
  gather_kernel<<<dim3(16, 8), 256, 0, stream>>>(n2, lists, cnt, padc, Ag);
  // 8) expert GEMM1 (fused silu -> bf16); static worst-case grid, early-exit
  gemm_bf16_kernel<1><<<dim3(32, 16, 8), 256, 0, stream>>>(
      Ag, 1024, w1, 4096, Hb, 4096, 1024, padc,
      1024L * 1024, 1024L * 4096, 1024L * 4096);
  // 9) expert GEMM2
  gemm_bf16_kernel<0><<<dim3(8, 16, 8), 256, 0, stream>>>(
      Hb, 4096, w2, 1024, Oe, 1024, 4096, padc,
      1024L * 4096, 4096L * 1024, 1024L * 1024);
  // 10) weighted scatter + residual -> output
  final_kernel<<<1024, 256, 0, stream>>>(xbuf, Oe, smap, swts, mlp_s,
                                         (float*)d_out);
}
